// InteractE_76579266887953
// MI455X (gfx1250) — matrix operand order repaired and checked by reading, not compiled, not run on hardware
//
#include <hip/hip_runtime.h>
#include <cstdint>
#include <cstddef>

// ---------------- problem constants ----------------
#define B_      1024
#define E_      200
#define Q_      10
#define NENT    100000
#define IPERM   2
#define CH      64
#define FS      9
#define HW20    20          // both spatial dims (2*IKW == IKH == 20)
#define NPOS    400         // 20*20
#define OC_     128         // CH*IPERM
#define FLAT_   51200       // 128*400
#define NFCP    208         // FC N padded (200 -> 13 tiles)
#define KFIN    224         // final-GEMM K padded (200 -> 7 k-steps of 32)
#define ALPHA_  0.8f
#define EPS_    1e-5f
#define KSPLIT  8
#define KCHUNK  (FLAT_ / KSPLIT)   // 6400
#define MT_     (B_ / 16)          // 64
#define NT_FC   (NFCP / 16)        // 13
#define NT_FIN  (NENT / 16)        // 6250
#define NGRP    5                  // n-tiles per wave in final GEMM
#define NGRPS   (NT_FIN / NGRP)    // 1250 n-groups

// ---------------- workspace layout (bytes) ---------
// total ~190.7 MB
#define O_X0     ((size_t)0)                       // 1024*800 f32        = 3,276,800
#define O_A0C0   ((size_t)3276800)                 // 4 f32 (+pad)
#define O_P1     ((size_t)3277056)                 // 128*8192*2 f32      = 8,388,608
#define O_A1C1   ((size_t)11665664)                // 256 f32
#define O_ACT1   ((size_t)11666688)                // 1024*51200 bf16     = 104,857,600
#define O_FCWT   ((size_t)116524288)               // 208*51200 bf16      = 21,299,200
#define O_FCPART ((size_t)137823488)               // 832*2048 f32        = 6,815,744
#define O_FCOUT  ((size_t)144639232)               // 1024*200 f32        = 819,200
#define O_A2C2   ((size_t)145458432)               // 400 f32 (+pad)
#define O_X2     ((size_t)145460224)               // 1024*224 bf16       = 458,752
#define O_ENTT   ((size_t)145918976)               // 100000*224 bf16     = 44,800,000

// ---------------- CDNA5 WMMA types -----------------
typedef __attribute__((ext_vector_type(16))) __bf16          v16bf;
typedef __attribute__((ext_vector_type(8)))  float           v8f;
typedef __attribute__((ext_vector_type(16))) unsigned short  v16u;
typedef __attribute__((ext_vector_type(8)))  unsigned short  v8u;

// ---------------- helpers --------------------------
__device__ __forceinline__ float bf16_to_f32(unsigned short h) {
  unsigned u = ((unsigned)h) << 16;
  float f; __builtin_memcpy(&f, &u, 4); return f;
}
__device__ __forceinline__ unsigned short f32_to_bf16(float f) {
  unsigned u; __builtin_memcpy(&u, &f, 4);
  unsigned r = u + 0x7FFFu + ((u >> 16) & 1u);     // round-to-nearest-even
  return (unsigned short)(r >> 16);
}
__device__ __forceinline__ float leaky02(float x) { return x > 0.f ? x : 0.2f * x; }
__device__ __forceinline__ int wrap20(int x) { return x < 0 ? x + 20 : (x >= 20 ? x - 20 : x); }

__device__ __forceinline__ float blk_sum(float v, float* red) {
  int t = threadIdx.x;
  red[t] = v; __syncthreads();
  for (int s = 128; s > 0; s >>= 1) { if (t < s) red[t] += red[t + s]; __syncthreads(); }
  float r = red[0]; __syncthreads();
  return r;
}

__device__ __forceinline__ v16bf pack_ab(v8u lo, v8u hi) {
  v16u u;
#pragma unroll
  for (int j = 0; j < 8; ++j) { u[j] = lo[j]; u[8 + j] = hi[j]; }
  return __builtin_bit_cast(v16bf, u);
}

// 16x16 bf16 WMMA tile K-loop.  Abase/Bbase already include the per-lane
// half-wave offsets (A: +hi*8, B: +hi*8).  Klen multiple of 32.
__device__ __forceinline__ v8f wmma_bf16_loop(const unsigned short* __restrict__ Abase,
                                              const unsigned short* __restrict__ Bbase,
                                              int Klen, v8f acc) {
  for (int kk = 0; kk < Klen; kk += 32) {
    v16bf av = pack_ab(*(const v8u*)(Abase + kk), *(const v8u*)(Abase + kk + 16));
    v16bf bv = pack_ab(*(const v8u*)(Bbase + kk), *(const v8u*)(Bbase + kk + 16));
    acc = __builtin_amdgcn_wmma_f32_16x16x32_bf16(false, av, false, bv,
                                                  (short)0, acc, false, false);
  }
  return acc;
}

// =========== K1: attention + combine + chequer gather -> x0 ===========
__global__ void k_attn(const int* __restrict__ sub, const int* __restrict__ rel,
                       const int* __restrict__ quals, const int* __restrict__ Nm,
                       const int* __restrict__ perm,
                       const float* __restrict__ ent, const float* __restrict__ relw,
                       const float* __restrict__ wsub, const float* __restrict__ wrel,
                       const float* __restrict__ wm, float* __restrict__ x0) {
  __shared__ float sSub[E_], sRel[E_], qr[Q_ * E_], qe[Q_ * E_];
  __shared__ float sa[E_], ra[E_], sC[2 * E_];
  __shared__ float scS[Q_], scR[Q_], f2[2], red[256];
  const int b = blockIdx.x, t = threadIdx.x;

  if (t < E_) {
    int si = sub[b];  sSub[t] = si ? ent[(size_t)si * E_ + t] : 0.f;
    int ri = rel[b];  sRel[t] = ri ? relw[(size_t)ri * E_ + t] : 0.f;
#pragma unroll
    for (int q = 0; q < Q_; ++q) {
      int rq = quals[b * 2 * Q_ + 2 * q];
      int eq = quals[b * 2 * Q_ + 2 * q + 1];
      qr[q * E_ + t] = rq ? relw[(size_t)rq * E_ + t] : 0.f;
      qe[q * E_ + t] = eq ? ent[(size_t)eq * E_ + t] : 0.f;
    }
  }
  __syncthreads();

  float tS = blk_sum(t < E_ ? sSub[t] * wsub[t] : 0.f, red);
  float tR = blk_sum(t < E_ ? sRel[t] * wrel[t] : 0.f, red);
  for (int q = 0; q < Q_; ++q) {
    float uS = blk_sum(t < E_ ? qr[q * E_ + t] * wsub[E_ + t] : 0.f, red);
    if (t == 0) scS[q] = tS + uS;
    float uR = blk_sum(t < E_ ? qr[q * E_ + t] * wrel[E_ + t] : 0.f, red);
    if (t == 0) scR[q] = tR + uR;
  }
  if (t == 0) {
    float tot = 0.f;
    for (int q = 0; q < Q_; ++q) {
      float w = (Nm[b * Q_ + q] != 0) ? __expf(-leaky02(scS[q])) : 0.f;
      scS[q] = w; tot += w;
    }
    if (tot == 0.f) tot = 1.f;
    for (int q = 0; q < Q_; ++q) scS[q] /= tot;
    tot = 0.f;
    for (int q = 0; q < Q_; ++q) {
      float w = (Nm[b * Q_ + q] != 0) ? __expf(-leaky02(scR[q])) : 0.f;
      scR[q] = w; tot += w;
    }
    if (tot == 0.f) tot = 1.f;
    for (int q = 0; q < Q_; ++q) scR[q] /= tot;
  }
  __syncthreads();

  if (t < E_) {
    float s = 0.f, r = 0.f;
#pragma unroll
    for (int q = 0; q < Q_; ++q) { float v = qe[q * E_ + t]; s += v * scS[q]; r += v * scR[q]; }
    sa[t] = s; ra[t] = r;
  }
  __syncthreads();

  float mS = blk_sum(t < E_ ? sa[t] * wm[t] : 0.f, red);
  float mR = blk_sum(t < E_ ? ra[t] * wm[t] : 0.f, red);
  if (t == 0) {
    float ss = __expf(-leaky02(mS)), rs = __expf(-leaky02(mR));
    float tt = ss + rs;
    f2[0] = ss / tt; f2[1] = rs / tt;
  }
  __syncthreads();

  if (t < E_) {
    float emb = f2[0] * sa[t] + f2[1] * ra[t];
    sC[t]       = sSub[t] * ALPHA_ + (1.f - ALPHA_) * emb;
    sC[E_ + t]  = sRel[t] * ALPHA_ + (1.f - ALPHA_) * emb;
  }
  __syncthreads();

  for (int pos = t; pos < IPERM * 2 * E_; pos += 256)
    x0[(size_t)b * 800 + pos] = sC[perm[pos]];
}

// =========== K2: BN0 stats -> scale/shift per IPERM channel ===========
__global__ void k_bn0(const float* __restrict__ x0, const float* __restrict__ g,
                      const float* __restrict__ bb, float* __restrict__ a0c0) {
  __shared__ float red[256], red2[256];
  const int ch = blockIdx.x, t = threadIdx.x;
  float s1 = 0.f, s2 = 0.f;
  for (int i = t; i < B_ * NPOS; i += 256) {
    int b = i / NPOS, p = i % NPOS;
    float v = x0[(size_t)b * 800 + ch * NPOS + p];
    s1 += v; s2 += v * v;
  }
  red[t] = s1; red2[t] = s2; __syncthreads();
  for (int s = 128; s > 0; s >>= 1) {
    if (t < s) { red[t] += red[t + s]; red2[t] += red2[t + s]; }
    __syncthreads();
  }
  if (t == 0) {
    float n = (float)(B_ * NPOS);
    float mean = red[0] / n, var = red2[0] / n - mean * mean;
    float a = g[ch] * rsqrtf(var + EPS_);
    a0c0[ch] = a; a0c0[2 + ch] = bb[ch] - mean * a;
  }
}

// =========== K3: grouped circular conv -> raw bf16 + per-oc partial stats ===========
__global__ void k_conv(const float* __restrict__ x0, const float* __restrict__ cf,
                       const float* __restrict__ a0c0,
                       unsigned short* __restrict__ act1, float* __restrict__ p1) {
  __shared__ float tile[800], filt[CH * FS * FS];
  const int b = blockIdx.x, t = threadIdx.x;
  const float a0 = a0c0[0], a1 = a0c0[1], c0 = a0c0[2], c1 = a0c0[3];
  for (int i = t; i < 800; i += 256) {
    float v = x0[(size_t)b * 800 + i];
    tile[i] = (i < 400) ? (a0 * v + c0) : (a1 * v + c1);
  }
  for (int i = t; i < CH * FS * FS; i += 256) filt[i] = cf[i];
  __syncthreads();

  const int lane = t & 31, wv = t >> 5;
  const bool act = (t < 200);
  const int h1 = t / HW20, w1 = t % HW20, h2 = h1 + 10;
  for (int oc = 0; oc < OC_; ++oc) {
    const float* F = filt + (oc & (CH - 1)) * (FS * FS);
    const float* X = tile + (oc >> 6) * NPOS;
    float y1 = 0.f, y2 = 0.f;
    if (act) {
#pragma unroll
      for (int i = 0; i < FS; ++i) {
        int r1 = wrap20(h1 + i - 4) * HW20;
        int r2 = wrap20(h2 + i - 4) * HW20;
#pragma unroll
        for (int j = 0; j < FS; ++j) {
          int c = wrap20(w1 + j - 4);
          float f = F[i * FS + j];
          y1 += X[r1 + c] * f;
          y2 += X[r2 + c] * f;
        }
      }
      size_t base = (size_t)b * FLAT_ + (size_t)oc * NPOS;
      act1[base + t]       = f32_to_bf16(y1);
      act1[base + t + 200] = f32_to_bf16(y2);
    }
    float s1 = y1 + y2, s2 = y1 * y1 + y2 * y2;
#pragma unroll
    for (int off = 16; off > 0; off >>= 1) {
      s1 += __shfl_down(s1, off);
      s2 += __shfl_down(s2, off);
    }
    if (lane == 0) {
      size_t idx = ((size_t)oc * 8192 + (size_t)b * 8 + wv) * 2;
      p1[idx] = s1; p1[idx + 1] = s2;
    }
  }
}

// =========== K4: finalize BN1 per out-channel ===========
__global__ void k_bn1fin(const float* __restrict__ p1, const float* __restrict__ g,
                         const float* __restrict__ bb, float* __restrict__ a1c1) {
  __shared__ float red[256], red2[256];
  const int oc = blockIdx.x, t = threadIdx.x;
  float s1 = 0.f, s2 = 0.f;
  for (int i = t; i < 8192; i += 256) {
    size_t idx = ((size_t)oc * 8192 + i) * 2;
    s1 += p1[idx]; s2 += p1[idx + 1];
  }
  red[t] = s1; red2[t] = s2; __syncthreads();
  for (int s = 128; s > 0; s >>= 1) {
    if (t < s) { red[t] += red[t + s]; red2[t] += red2[t + s]; }
    __syncthreads();
  }
  if (t == 0) {
    float n = (float)(B_ * NPOS);
    float mean = red[0] / n, var = red2[0] / n - mean * mean;
    float a = g[oc] * rsqrtf(var + EPS_);
    a1c1[oc] = a; a1c1[OC_ + oc] = bb[oc] - mean * a;
  }
}

// =========== K5: BN1 + ReLU in place (bf16) ===========
__global__ void k_bnrelu(unsigned short* __restrict__ act1, const float* __restrict__ a1c1) {
  size_t idx = (size_t)blockIdx.x * 256 + threadIdx.x;
  int oc = (int)((idx / NPOS) % OC_);
  float v = bf16_to_f32(act1[idx]);
  float y = a1c1[oc] * v + a1c1[OC_ + oc];
  act1[idx] = f32_to_bf16(y > 0.f ? y : 0.f);
}

// =========== K6: fc_w -> bf16 transposed [n][k] (n padded to 208) ===========
__global__ void k_fcw(const float* __restrict__ fcw, unsigned short* __restrict__ fcwT) {
  size_t idx = (size_t)blockIdx.x * 256 + threadIdx.x;   // over 208*51200, k-major read
  int k = (int)(idx / NFCP), n = (int)(idx % NFCP);
  float v = (n < E_) ? fcw[(size_t)k * E_ + n] : 0.f;
  fcwT[(size_t)n * FLAT_ + k] = f32_to_bf16(v);
}

// =========== K7: FC GEMM (WMMA bf16, K split 8 ways) ===========
__global__ void k_fcgemm(const unsigned short* __restrict__ act1,
                         const unsigned short* __restrict__ fcwT,
                         float* __restrict__ part) {
  const int t = threadIdx.x, lane = t & 31, wv = t >> 5;
  const int tile = blockIdx.x;                 // 0..831
  const int s = wv;                            // k-chunk
  const int m0 = (tile / NT_FC) * 16, n0 = (tile % NT_FC) * 16;
  const int mrow = lane & 15, hi = lane >> 4;
  const unsigned short* A = act1 + (size_t)(m0 + mrow) * FLAT_ + (size_t)s * KCHUNK + hi * 8;
  const unsigned short* Bp = fcwT + (size_t)(n0 + mrow) * FLAT_ + (size_t)s * KCHUNK + hi * 8;
  v8f acc = {};
  acc = wmma_bf16_loop(A, Bp, KCHUNK, acc);
  size_t base = (size_t)tile * 2048 + (size_t)s * 256;
#pragma unroll
  for (int r = 0; r < 8; ++r) part[base + r * 32 + lane] = acc[r];
}

// =========== K8: combine K-split partials + fc bias -> fc_out f32 ===========
__global__ void k_fccomb(const float* __restrict__ part, const float* __restrict__ fcb,
                         float* __restrict__ fcout) {
  const int t = threadIdx.x, tile = blockIdx.x;
  const int lane = t & 31, r = t >> 5;
  const int m0 = (tile / NT_FC) * 16, n0 = (tile % NT_FC) * 16;
  float v = 0.f;
#pragma unroll
  for (int s = 0; s < KSPLIT; ++s)
    v += part[(size_t)tile * 2048 + (size_t)s * 256 + r * 32 + lane];
  int row = m0 + r + (lane >> 4) * 8;
  int col = n0 + (lane & 15);
  if (col < E_) fcout[(size_t)row * E_ + col] = v + fcb[col];
}

// =========== K9: BN2 stats -> scale/shift per feature ===========
__global__ void k_bn2(const float* __restrict__ fcout, const float* __restrict__ g,
                      const float* __restrict__ bb, float* __restrict__ a2c2) {
  __shared__ float red[256], red2[256];
  const int e = blockIdx.x, t = threadIdx.x;
  float s1 = 0.f, s2 = 0.f;
  for (int b = t; b < B_; b += 256) {
    float v = fcout[(size_t)b * E_ + e];
    s1 += v; s2 += v * v;
  }
  red[t] = s1; red2[t] = s2; __syncthreads();
  for (int s = 128; s > 0; s >>= 1) {
    if (t < s) { red[t] += red[t + s]; red2[t] += red2[t + s]; }
    __syncthreads();
  }
  if (t == 0) {
    float n = (float)B_;
    float mean = red[0] / n, var = red2[0] / n - mean * mean;
    float a = g[e] * rsqrtf(var + EPS_);
    a2c2[e] = a; a2c2[E_ + e] = bb[e] - mean * a;
  }
}

// =========== K10: BN2 + ReLU -> bf16 X2 (K padded to 224) ===========
__global__ void k_x2(const float* __restrict__ fcout, const float* __restrict__ a2c2,
                     unsigned short* __restrict__ x2) {
  size_t idx = (size_t)blockIdx.x * 256 + threadIdx.x;   // over 1024*224
  int b = (int)(idx / KFIN), k = (int)(idx % KFIN);
  float y = 0.f;
  if (k < E_) {
    float v = fcout[(size_t)b * E_ + k];
    y = a2c2[k] * v + a2c2[E_ + k];
    if (y < 0.f) y = 0.f;
  }
  x2[idx] = f32_to_bf16(y);
}

// =========== K11: ent_embed -> bf16 [n][224], row 0 zeroed ===========
__global__ void k_entT(const float* __restrict__ ent, unsigned short* __restrict__ entT) {
  size_t idx = (size_t)blockIdx.x * 256 + threadIdx.x;   // over 100000*224
  int n = (int)(idx / KFIN), k = (int)(idx % KFIN);
  float v = (k < E_ && n != 0) ? ent[(size_t)n * E_ + k] : 0.f;
  entT[idx] = f32_to_bf16(v);
}

// =========== K12: final GEMM (WMMA bf16), 1 m-tile x 5 n-tiles per wave ===========
// A register tile is reused across 5 accumulators: 1.2KB loads per wmma
// instead of 2KB, and X2 L2 traffic drops 5x.
__global__ void k_final(const unsigned short* __restrict__ x2,
                        const unsigned short* __restrict__ entT,
                        const float* __restrict__ bias, float* __restrict__ out) {
  const int t = threadIdx.x, lane = t & 31, wv = t >> 5;
  const int gid = blockIdx.x * 8 + wv;            // 0..79999
  const int mIdx = gid / NGRPS;                   // 0..63
  const int ngrp = gid % NGRPS;                   // 0..1249
  const int m0 = mIdx * 16, n0 = ngrp * (16 * NGRP);
  const int mrow = lane & 15, hi = lane >> 4;

  const unsigned short* A = x2 + (size_t)(m0 + mrow) * KFIN + hi * 8;
  const unsigned short* Bp[NGRP];
#pragma unroll
  for (int g = 0; g < NGRP; ++g)
    Bp[g] = entT + (size_t)(n0 + g * 16 + mrow) * KFIN + hi * 8;

  v8f acc[NGRP];
#pragma unroll
  for (int g = 0; g < NGRP; ++g) acc[g] = (v8f){};

#pragma unroll
  for (int kk = 0; kk < KFIN; kk += 32) {
    v16bf av = pack_ab(*(const v8u*)(A + kk), *(const v8u*)(A + kk + 16));
#pragma unroll
    for (int g = 0; g < NGRP; ++g) {
      v16bf bv = pack_ab(*(const v8u*)(Bp[g] + kk), *(const v8u*)(Bp[g] + kk + 16));
      acc[g] = __builtin_amdgcn_wmma_f32_16x16x32_bf16(false, av, false, bv,
                                                       (short)0, acc[g], false, false);
    }
  }

#pragma unroll
  for (int g = 0; g < NGRP; ++g) {
    const int col = n0 + g * 16 + (lane & 15);
    const float bv = bias[col];
#pragma unroll
    for (int r = 0; r < 8; ++r) {
      int row = m0 + r + (lane >> 4) * 8;
      float v = acc[g][r] + bv;
      out[(size_t)row * NENT + col] = 1.f / (1.f + __expf(-v));
    }
  }
}

// ================= host launcher =================
extern "C" void kernel_launch(void* const* d_in, const int* in_sizes, int n_in,
                              void* d_out, int out_size, void* d_ws, size_t ws_size,
                              hipStream_t stream) {
  const int*   sub   = (const int*)d_in[0];
  const int*   rel   = (const int*)d_in[1];
  const int*   quals = (const int*)d_in[2];
  const int*   Nm    = (const int*)d_in[3];
  const int*   perm  = (const int*)d_in[4];
  const float* ent   = (const float*)d_in[5];
  const float* relw  = (const float*)d_in[6];
  const float* wsub  = (const float*)d_in[7];
  const float* wrel  = (const float*)d_in[8];
  const float* wm    = (const float*)d_in[9];
  const float* cf    = (const float*)d_in[10];
  const float* fcw   = (const float*)d_in[11];
  const float* fcb   = (const float*)d_in[12];
  const float* bias  = (const float*)d_in[13];
  const float* bn0g  = (const float*)d_in[14];
  const float* bn0b  = (const float*)d_in[15];
  const float* bn1g  = (const float*)d_in[16];
  const float* bn1b  = (const float*)d_in[17];
  const float* bn2g  = (const float*)d_in[18];
  const float* bn2b  = (const float*)d_in[19];
  float* out = (float*)d_out;

  char* ws = (char*)d_ws;
  float*          x0    = (float*)(ws + O_X0);
  float*          a0c0  = (float*)(ws + O_A0C0);
  float*          p1    = (float*)(ws + O_P1);
  float*          a1c1  = (float*)(ws + O_A1C1);
  unsigned short* act1  = (unsigned short*)(ws + O_ACT1);
  unsigned short* fcwT  = (unsigned short*)(ws + O_FCWT);
  float*          part  = (float*)(ws + O_FCPART);
  float*          fcout = (float*)(ws + O_FCOUT);
  float*          a2c2  = (float*)(ws + O_A2C2);
  unsigned short* x2    = (unsigned short*)(ws + O_X2);
  unsigned short* entT  = (unsigned short*)(ws + O_ENTT);

  k_attn  <<<B_,                    256, 0, stream>>>(sub, rel, quals, Nm, perm, ent, relw, wsub, wrel, wm, x0);
  k_bn0   <<<IPERM,                 256, 0, stream>>>(x0, bn0g, bn0b, a0c0);
  k_conv  <<<B_,                    256, 0, stream>>>(x0, cf, a0c0, act1, p1);
  k_bn1fin<<<OC_,                   256, 0, stream>>>(p1, bn1g, bn1b, a1c1);
  k_bnrelu<<<(B_ * FLAT_) / 256,    256, 0, stream>>>(act1, a1c1);
  k_fcw   <<<(NFCP * FLAT_) / 256,  256, 0, stream>>>(fcw, fcwT);
  k_fcgemm<<<MT_ * NT_FC,           256, 0, stream>>>(act1, fcwT, part);
  k_fccomb<<<MT_ * NT_FC,           256, 0, stream>>>(part, fcb, fcout);
  k_bn2   <<<E_,                    256, 0, stream>>>(fcout, bn2g, bn2b, a2c2);
  k_x2    <<<(B_ * KFIN) / 256,     256, 0, stream>>>(fcout, a2c2, x2);
  k_entT  <<<(NENT * KFIN) / 256,   256, 0, stream>>>(ent, entT);
  k_final <<<(MT_ * NGRPS) / 8,     256, 0, stream>>>(x2, entT, bias, out);
}
